// MultiGuideMono_22282290331726
// MI455X (gfx1250) — matrix operand order repaired and checked by reading, not compiled, not run on hardware
//
#include <hip/hip_runtime.h>
#include <hip/hip_bf16.h>

typedef _Float16 v16h __attribute__((ext_vector_type(16)));
typedef _Float16 v8h  __attribute__((ext_vector_type(8)));
typedef float    v8f  __attribute__((ext_vector_type(8)));

#define DEV static __device__ __forceinline__

DEV v16h mk16(v8h lo, v8h hi) {
    v16h r;
#pragma unroll
    for (int i = 0; i < 8; ++i) { r[i] = lo[i]; r[i + 8] = hi[i]; }
    return r;
}

// 16 contiguous f16 (two 16B loads)
DEV v16h ld16(const _Float16* p) {
    v8h lo = *(const v8h*)p;
    v8h hi = *(const v8h*)(p + 8);
    return mk16(lo, hi);
}

DEV v8f wmma16(v16h a, v16h b, v8f c) {
    // D(16x16 f32) = A(16x32 f16) * B(32x16 f16) + C
    return __builtin_amdgcn_wmma_f32_16x16x32_f16(false, a, false, b, (short)0, c, false, false);
}

// ---------------------------------------------------------------------------
// geometry
// ---------------------------------------------------------------------------
#define CCH   32
#define H0    256
#define W0    512
#define HW0   (H0*W0)        // 131072
#define H1    128
#define W1    256
#define H2    64
#define W2    128
#define NSEQ  (H2*W2)        // 8192
#define NB    2

// workspace offsets (bytes)
#define WS_MONOH  ((size_t)0)                         // 2*131072*32*2 = 16 MB
#define WS_COSTH  ((size_t)16777216)                  // 16 MB
#define WS_WF     ((size_t)33554432)                  // 6*9*32*32*2 = 110592
#define WS_QKVW   ((size_t)33665024)                  // 3*32*32*2 = 6144
#define WS_ME1    ((size_t)33671168)                  // 2*128*256*32*2 = 4 MB
#define WS_XE1    ((size_t)37865472)                  // 4 MB
#define WS_MONOF  ((size_t)42059776)                  // 2*8192*32*2 = 1 MB
#define WS_MULTF  ((size_t)43108352)                  // 1 MB
#define WS_RG1    ((size_t)44156928)                  // 16 MB
#define WS_QH     ((size_t)60934144)                  // 1 MB
#define WS_KH     ((size_t)61982720)                  // 1 MB
#define WS_VT     ((size_t)63031296)                  // 1 MB
#define WS_MOUT   ((size_t)64079872)                  // 2*32*8192*4 = 2 MB

// ---------------------------------------------------------------------------
// prep: conv weights OIHW f32 -> [tap][oc][ic] f16 ; qkv weights -> f16
// ---------------------------------------------------------------------------
__global__ void prep_weights(const float* __restrict__ w0, const float* __restrict__ w1,
                             const float* __restrict__ w2, const float* __restrict__ w3,
                             const float* __restrict__ w4, const float* __restrict__ w5,
                             const float* __restrict__ qw, const float* __restrict__ kw,
                             const float* __restrict__ vw,
                             _Float16* __restrict__ wf, _Float16* __restrict__ qkvf) {
    int i = blockIdx.x * blockDim.x + threadIdx.x;
    const int NCONV = 6 * 9 * 32 * 32;
    if (i < NCONV) {
        int layer = i / 9216;
        int rem   = i - layer * 9216;
        int tap   = rem >> 10;
        int oc    = (rem >> 5) & 31;
        int ic    = rem & 31;
        int ky = tap / 3, kx = tap - ky * 3;
        const float* ws[6] = { w0, w1, w2, w3, w4, w5 };
        wf[i] = (_Float16)ws[layer][((oc * 32 + ic) * 3 + ky) * 3 + kx];
    } else {
        int j = i - NCONV;
        if (j < 3 * 1024) {
            const float* qs[3] = { qw, kw, vw };
            qkvf[j] = (_Float16)qs[j >> 10][j & 1023];   // [oc][ic] row-major
        }
    }
}

// NCHW f32 -> [b][p][c] f16
__global__ void nchw_to_chlast(const float* __restrict__ x, _Float16* __restrict__ out, int HW) {
    int i = blockIdx.x * blockDim.x + threadIdx.x;
    if (i >= NB * CCH * HW) return;
    int c = i & 31;
    int rest = i >> 5;
    int p = rest % HW;
    int b = rest / HW;
    out[i] = (_Float16)x[((size_t)b * CCH + c) * HW + p];
}

// ---------------------------------------------------------------------------
// 3x3 implicit-GEMM tap accumulation. CHECK=false: interior tile, straight-line
// unconditional loads (no exec masking); CHECK=true: zero-pad boundary tile.
// ---------------------------------------------------------------------------
template <bool CHECK>
DEV void conv_taps(const _Float16* __restrict__ inb, const _Float16* __restrict__ wf,
                   int y, int xb, int m, bool hl, int Hin, int Win, int stride,
                   v8f& acc0, v8f& acc1) {
    int ca0 = hl ? 0 : 16, ca1 = hl ? 8 : 24;   // A-frag channel chunks
    int cw  = hl ? 0 : 16;                       // B-frag ic chunk
    int xi0 = (xb + m) * stride - 1;
#pragma unroll
    for (int ky = 0; ky < 3; ++ky) {
        int yi = y * stride + ky - 1;
        bool yok = !CHECK || ((unsigned)yi < (unsigned)Hin);
#pragma unroll
        for (int kx = 0; kx < 3; ++kx) {
            int xi = xi0 + kx;
            v16h a;
            if (CHECK) {
                v8h alo = {}, ahi = {};
                if (yok && (unsigned)xi < (unsigned)Win) {
                    const _Float16* ap = inb + ((size_t)yi * Win + xi) * CCH;
                    alo = *(const v8h*)(ap + ca0);
                    ahi = *(const v8h*)(ap + ca1);
                }
                a = mk16(alo, ahi);
            } else {
                const _Float16* ap = inb + ((size_t)yi * Win + xi) * CCH;
                a = mk16(*(const v8h*)(ap + ca0), *(const v8h*)(ap + ca1));
            }
            int tap = ky * 3 + kx;
            v16h bf0 = ld16(wf + (size_t)(tap * 32 + m)      * 32 + cw);
            v16h bf1 = ld16(wf + (size_t)(tap * 32 + m + 16) * 32 + cw);
            acc0 = wmma16(a, bf0, acc0);
            acc1 = wmma16(a, bf1, acc1);
        }
    }
}

// ---------------------------------------------------------------------------
// 3x3 conv + bias + BN + ReLU, implicit GEMM via WMMA, f16 channel-last out.
// One wave = 16 output pixels along x (same y), all 32 output channels.
// ---------------------------------------------------------------------------
__global__ void conv3x3_bn_relu_f16(const _Float16* __restrict__ in,
                                    _Float16* __restrict__ out,
                                    const _Float16* __restrict__ wf,   // [9][32oc][32ic] f16
                                    const float* __restrict__ cb,
                                    const float* __restrict__ bg,
                                    const float* __restrict__ bb,
                                    int Hin, int Win, int Hout, int Wout,
                                    int stride, int nwave) {
    int wave = blockIdx.x * (blockDim.x >> 5) + (threadIdx.x >> 5);
    if (wave >= nwave) return;
    int tiles_x = Wout >> 4;
    int tpb = Hout * tiles_x;
    int b   = wave / tpb;
    int rem = wave - b * tpb;
    int y   = rem / tiles_x;
    int xb  = (rem - y * tiles_x) << 4;

    int lane = threadIdx.x & 31;
    int m    = lane & 15;
    bool hl  = lane < 16;

    const _Float16* inb = in + (size_t)b * Hin * Win * CCH;
    v8f acc0 = {}, acc1 = {};

    // wave-uniform interior test: all 9 taps of all 16 pixels in-bounds
    bool interior = (y * stride >= 1) && (y * stride + 1 < Hin) &&
                    (xb * stride >= 1) && ((xb + 15) * stride + 1 < Win);
    if (interior)
        conv_taps<false>(inb, wf, y, xb, m, hl, Hin, Win, stride, acc0, acc1);
    else
        conv_taps<true>(inb, wf, y, xb, m, hl, Hin, Win, stride, acc0, acc1);

    float rs  = rsqrtf(1.0f + 1e-5f);
    int oc0 = m, oc1 = m + 16;
    float bi0 = cb[oc0], sc0 = bg[oc0] * rs, be0 = bb[oc0];
    float bi1 = cb[oc1], sc1 = bg[oc1] * rs, be1 = bb[oc1];

    _Float16* ob = out + (size_t)b * Hout * Wout * CCH;
    int roff = hl ? 0 : 8;
#pragma unroll
    for (int r = 0; r < 8; ++r) {
        int px = xb + r + roff;
        float v0 = fmaxf((acc0[r] + bi0) * sc0 + be0, 0.0f);
        float v1 = fmaxf((acc1[r] + bi1) * sc1 + be1, 0.0f);
        _Float16* op = ob + ((size_t)y * Wout + px) * CCH;
        op[oc0] = (_Float16)v0;
        op[oc1] = (_Float16)v1;
    }
}

// ---------------------------------------------------------------------------
// q/k/v 1x1 convs (32x32 matmuls) via WMMA.
// Qh,Kh: [b][p][32] f16 (row-major, A/B-frag ready), Vt: [b][32][p] f16.
// ---------------------------------------------------------------------------
__global__ void qkv_kernel(const _Float16* __restrict__ multif,
                           const _Float16* __restrict__ monof,
                           const _Float16* __restrict__ qkvw,  // [3][32oc][32ic]
                           const float* __restrict__ qb, const float* __restrict__ kb,
                           const float* __restrict__ vb,
                           _Float16* __restrict__ Qh, _Float16* __restrict__ Kh,
                           _Float16* __restrict__ Vt) {
    int wave = blockIdx.x * (blockDim.x >> 5) + (threadIdx.x >> 5);
    if (wave >= NB * (NSEQ / 16)) return;
    int b = wave >> 9;          // 512 tiles per batch
    int t = wave & 511;
    int p0 = t << 4;

    int lane = threadIdx.x & 31;
    int m = lane & 15;
    bool hl = lane < 16;
    int ca0 = hl ? 0 : 16, ca1 = hl ? 8 : 24;
    int cw  = hl ? 0 : 16;

    const _Float16* am = multif + ((size_t)b * NSEQ + p0 + m) * CCH;
    const _Float16* av = monof  + ((size_t)b * NSEQ + p0 + m) * CCH;
    v16h A_multi = mk16(*(const v8h*)(am + ca0), *(const v8h*)(am + ca1));
    v16h A_mono  = mk16(*(const v8h*)(av + ca0), *(const v8h*)(av + ca1));

    const _Float16* qw = qkvw;
    const _Float16* kw = qkvw + 1024;
    const _Float16* vw = qkvw + 2048;
    v16h qB0 = ld16(qw + (m)      * 32 + cw), qB1 = ld16(qw + (m + 16) * 32 + cw);
    v16h kB0 = ld16(kw + (m)      * 32 + cw), kB1 = ld16(kw + (m + 16) * 32 + cw);
    v16h vB0 = ld16(vw + (m)      * 32 + cw), vB1 = ld16(vw + (m + 16) * 32 + cw);

    v8f z = {};
    v8f q0 = wmma16(A_multi, qB0, z), q1 = wmma16(A_multi, qB1, z);
    v8f k0 = wmma16(A_multi, kB0, z), k1 = wmma16(A_multi, kB1, z);
    v8f v0 = wmma16(A_mono,  vB0, z), v1 = wmma16(A_mono,  vB1, z);

    int oc0 = m, oc1 = m + 16;
    float qb0 = qb[oc0], qb1 = qb[oc1];
    float kb0 = kb[oc0], kb1 = kb[oc1];
    float vb0 = vb[oc0], vb1 = vb[oc1];
    int roff = hl ? 0 : 8;
#pragma unroll
    for (int r = 0; r < 8; ++r) {
        int p = p0 + r + roff;
        size_t base = ((size_t)b * NSEQ + p) * CCH;
        Qh[base + oc0] = (_Float16)(q0[r] + qb0);
        Qh[base + oc1] = (_Float16)(q1[r] + qb1);
        Kh[base + oc0] = (_Float16)(k0[r] + kb0);
        Kh[base + oc1] = (_Float16)(k1[r] + kb1);
        Vt[((size_t)b * CCH + oc0) * NSEQ + p] = (_Float16)(v0[r] + vb0);
        Vt[((size_t)b * CCH + oc1) * NSEQ + p] = (_Float16)(v1[r] + vb1);
    }
}

// ---------------------------------------------------------------------------
// Flash attention: one wave = 16 queries, streams 8192 keys in blocks of 32.
// mono_out: [b][32][8192] f32
// ---------------------------------------------------------------------------
__global__ void attn_kernel(const _Float16* __restrict__ Qh,
                            const _Float16* __restrict__ Kh,
                            const _Float16* __restrict__ Vt,
                            float* __restrict__ mono_out) {
    __shared__ _Float16 lds_p[8][16][32];   // per-wave P staging (C-layout -> A-layout)
    int wave = blockIdx.x * (blockDim.x >> 5) + (threadIdx.x >> 5);
    if (wave >= NB * (NSEQ / 16)) return;
    int lw = (threadIdx.x >> 5) & 7;
    int b  = wave >> 9;
    int qt = wave & 511;

    int lane = threadIdx.x & 31;
    int m = lane & 15;
    bool hl = lane < 16;
    int roff = hl ? 0 : 8;

    const _Float16* Qb = Qh + (size_t)b * NSEQ * CCH;
    const _Float16* Kb = Kh + (size_t)b * NSEQ * CCH;
    const _Float16* Vb = Vt + (size_t)b * CCH * NSEQ;

    // Q A-fragment (16 queries x 32 ch)
    const _Float16* qrow = Qb + ((size_t)qt * 16 + m) * CCH;
    v16h qa = mk16(*(const v8h*)(qrow + (hl ? 0 : 16)), *(const v8h*)(qrow + (hl ? 8 : 24)));

    v8f o0 = {}, o1 = {};
    float mrow[8], lrow[8];
#pragma unroll
    for (int r = 0; r < 8; ++r) { mrow[r] = -1e30f; lrow[r] = 0.0f; }

    for (int j = 0; j < NSEQ; j += 32) {
        // K B-fragments: keys j..j+15 and j+16..j+31 (K dim = 32 channels)
        v16h kf0 = ld16(Kb + ((size_t)(j + m)      * CCH) + (hl ? 0 : 16));
        v16h kf1 = ld16(Kb + ((size_t)(j + 16 + m) * CCH) + (hl ? 0 : 16));

        // prefetch next key block (global_prefetch_b8); harmless past-end within ws
        __builtin_prefetch(Kb + (size_t)(j + 32 + m) * CCH, 0, 0);
        __builtin_prefetch(Vb + (size_t)m * NSEQ + j + 32, 0, 0);

        v8f z = {};
        v8f s0 = wmma16(qa, kf0, z);
        v8f s1 = wmma16(qa, kf1, z);

        // row max over the 32 keys (butterfly within 16-lane halves)
        float pm[8];
#pragma unroll
        for (int r = 0; r < 8; ++r) pm[r] = fmaxf(s0[r], s1[r]);
#pragma unroll
        for (int mask = 1; mask < 16; mask <<= 1)
#pragma unroll
            for (int r = 0; r < 8; ++r) pm[r] = fmaxf(pm[r], __shfl_xor(pm[r], mask, 32));

        float alpha[8], ps[8];
#pragma unroll
        for (int r = 0; r < 8; ++r) {
            float mn = fmaxf(mrow[r], pm[r]);
            alpha[r] = __expf(mrow[r] - mn);
            mrow[r]  = mn;
            float p0 = __expf(s0[r] - mn);
            float p1 = __expf(s1[r] - mn);
            s0[r] = p0; s1[r] = p1;
            ps[r] = p0 + p1;
        }
#pragma unroll
        for (int mask = 1; mask < 16; mask <<= 1)
#pragma unroll
            for (int r = 0; r < 8; ++r) ps[r] += __shfl_xor(ps[r], mask, 32);
#pragma unroll
        for (int r = 0; r < 8; ++r) {
            lrow[r] = lrow[r] * alpha[r] + ps[r];
            o0[r] *= alpha[r];
            o1[r] *= alpha[r];
        }

        // P: C-layout -> A-layout through LDS
#pragma unroll
        for (int r = 0; r < 8; ++r) {
            lds_p[lw][r + roff][m]      = (_Float16)s0[r];
            lds_p[lw][r + roff][16 + m] = (_Float16)s1[r];
        }
        __asm__ volatile("s_wait_dscnt 0" ::: "memory");
        const _Float16* prow = &lds_p[lw][m][0];
        v16h pa = mk16(*(const v8h*)(prow + (hl ? 0 : 16)), *(const v8h*)(prow + (hl ? 8 : 24)));

        // V B-fragments: channels m / m+16, K dim = 32 keys of this block
        v16h vf0 = ld16(Vb + (size_t)m        * NSEQ + j + (hl ? 0 : 16));
        v16h vf1 = ld16(Vb + (size_t)(m + 16) * NSEQ + j + (hl ? 0 : 16));
        o0 = wmma16(pa, vf0, o0);
        o1 = wmma16(pa, vf1, o1);
    }

    float* ob = mono_out + (size_t)b * CCH * NSEQ;
#pragma unroll
    for (int r = 0; r < 8; ++r) {
        float inv = 1.0f / lrow[r];
        int p = qt * 16 + r + roff;
        ob[(size_t)m        * NSEQ + p] = o0[r] * inv;
        ob[(size_t)(m + 16) * NSEQ + p] = o1[r] * inv;
    }
}

// ---------------------------------------------------------------------------
// rg conv2 (stride 1, 256x512) fused with  out = res + gamma * upsample4(mono_out)
// writes NCHW f32 to d_out
// ---------------------------------------------------------------------------
__global__ void conv3x3_fused_out(const _Float16* __restrict__ in,
                                  float* __restrict__ out,
                                  const _Float16* __restrict__ wf,
                                  const float* __restrict__ cb,
                                  const float* __restrict__ bg,
                                  const float* __restrict__ bb,
                                  const float* __restrict__ mono_out,
                                  const float* __restrict__ gamma) {
    int wave = blockIdx.x * (blockDim.x >> 5) + (threadIdx.x >> 5);
    const int tiles_x = W0 >> 4;               // 32
    const int tpb = H0 * tiles_x;              // 8192
    if (wave >= NB * tpb) return;
    int b   = wave / tpb;
    int rem = wave - b * tpb;
    int y   = rem / tiles_x;
    int xb  = (rem - y * tiles_x) << 4;

    int lane = threadIdx.x & 31;
    int m = lane & 15;
    bool hl = lane < 16;

    const _Float16* inb = in + (size_t)b * HW0 * CCH;
    v8f acc0 = {}, acc1 = {};

    bool interior = (y >= 1) && (y + 1 < H0) && (xb >= 1) && (xb + 16 < W0);
    if (interior)
        conv_taps<false>(inb, wf, y, xb, m, hl, H0, W0, 1, acc0, acc1);
    else
        conv_taps<true>(inb, wf, y, xb, m, hl, H0, W0, 1, acc0, acc1);

    float rs = rsqrtf(1.0f + 1e-5f);
    float g0 = gamma[0];
    int oc0 = m, oc1 = m + 16;
    float bi0 = cb[oc0], sc0 = bg[oc0] * rs, be0 = bb[oc0];
    float bi1 = cb[oc1], sc1 = bg[oc1] * rs, be1 = bb[oc1];

    int roff = hl ? 0 : 8;
#pragma unroll
    for (int r = 0; r < 8; ++r) {
        int px = xb + r + roff;
        int upi = (y >> 2) * W2 + (px >> 2);
        float v0 = fmaxf((acc0[r] + bi0) * sc0 + be0, 0.0f)
                 + g0 * mono_out[((size_t)b * CCH + oc0) * NSEQ + upi];
        float v1 = fmaxf((acc1[r] + bi1) * sc1 + be1, 0.0f)
                 + g0 * mono_out[((size_t)b * CCH + oc1) * NSEQ + upi];
        out[(((size_t)b * CCH + oc0) * H0 + y) * W0 + px] = v0;
        out[(((size_t)b * CCH + oc1) * H0 + y) * W0 + px] = v1;
    }
}

// ---------------------------------------------------------------------------
extern "C" void kernel_launch(void* const* d_in, const int* in_sizes, int n_in,
                              void* d_out, int out_size, void* d_ws, size_t ws_size,
                              hipStream_t stream) {
    const float* mono = (const float*)d_in[0];
    const float* cost = (const float*)d_in[1];
    const float *me_w1 = (const float*)d_in[2],  *me_b1 = (const float*)d_in[3],
                *me_g1 = (const float*)d_in[4],  *me_be1 = (const float*)d_in[5];
    const float *me_w2 = (const float*)d_in[6],  *me_b2 = (const float*)d_in[7],
                *me_g2 = (const float*)d_in[8],  *me_be2 = (const float*)d_in[9];
    const float *xe_w1 = (const float*)d_in[10], *xe_b1 = (const float*)d_in[11],
                *xe_g1 = (const float*)d_in[12], *xe_be1 = (const float*)d_in[13];
    const float *xe_w2 = (const float*)d_in[14], *xe_b2 = (const float*)d_in[15],
                *xe_g2 = (const float*)d_in[16], *xe_be2 = (const float*)d_in[17];
    const float *rg_w1 = (const float*)d_in[18], *rg_b1 = (const float*)d_in[19],
                *rg_g1 = (const float*)d_in[20], *rg_be1 = (const float*)d_in[21];
    const float *rg_w2 = (const float*)d_in[22], *rg_b2 = (const float*)d_in[23],
                *rg_g2 = (const float*)d_in[24], *rg_be2 = (const float*)d_in[25];
    const float *q_w = (const float*)d_in[26], *q_b = (const float*)d_in[27];
    const float *k_w = (const float*)d_in[28], *k_b = (const float*)d_in[29];
    const float *v_w = (const float*)d_in[30], *v_b = (const float*)d_in[31];
    const float *gamma = (const float*)d_in[32];

    char* ws = (char*)d_ws;
    _Float16* monoh  = (_Float16*)(ws + WS_MONOH);
    _Float16* costh  = (_Float16*)(ws + WS_COSTH);
    _Float16* wf     = (_Float16*)(ws + WS_WF);
    _Float16* qkvf   = (_Float16*)(ws + WS_QKVW);
    _Float16* me1b   = (_Float16*)(ws + WS_ME1);
    _Float16* xe1b   = (_Float16*)(ws + WS_XE1);
    _Float16* monof  = (_Float16*)(ws + WS_MONOF);
    _Float16* multif = (_Float16*)(ws + WS_MULTF);
    _Float16* rg1b   = (_Float16*)(ws + WS_RG1);
    _Float16* Qhb    = (_Float16*)(ws + WS_QH);
    _Float16* Khb    = (_Float16*)(ws + WS_KH);
    _Float16* Vtb    = (_Float16*)(ws + WS_VT);
    float*    moutb  = (float*)  (ws + WS_MOUT);
    float*    outp   = (float*)d_out;

    // 1. weight + activation preparation
    prep_weights<<<(6 * 9216 + 3 * 1024 + 255) / 256, 256, 0, stream>>>(
        me_w1, me_w2, xe_w1, xe_w2, rg_w1, rg_w2, q_w, k_w, v_w, wf, qkvf);
    nchw_to_chlast<<<(NB * CCH * HW0 + 255) / 256, 256, 0, stream>>>(mono, monoh, HW0);
    nchw_to_chlast<<<(NB * CCH * HW0 + 255) / 256, 256, 0, stream>>>(cost, costh, HW0);

    // 2. expand paths (stride-2 convs)
    conv3x3_bn_relu_f16<<<512, 256, 0, stream>>>(monoh, me1b, wf + 0 * 9216,
        me_b1, me_g1, me_be1, H0, W0, H1, W1, 2, NB * H1 * (W1 / 16));
    conv3x3_bn_relu_f16<<<128, 256, 0, stream>>>(me1b, monof, wf + 1 * 9216,
        me_b2, me_g2, me_be2, H1, W1, H2, W2, 2, NB * H2 * (W2 / 16));
    conv3x3_bn_relu_f16<<<512, 256, 0, stream>>>(costh, xe1b, wf + 2 * 9216,
        xe_b1, xe_g1, xe_be1, H0, W0, H1, W1, 2, NB * H1 * (W1 / 16));
    conv3x3_bn_relu_f16<<<128, 256, 0, stream>>>(xe1b, multif, wf + 3 * 9216,
        xe_b2, xe_g2, xe_be2, H1, W1, H2, W2, 2, NB * H2 * (W2 / 16));

    // 3. residual guidance conv1 (stride 1)
    conv3x3_bn_relu_f16<<<2048, 256, 0, stream>>>(monoh, rg1b, wf + 4 * 9216,
        rg_b1, rg_g1, rg_be1, H0, W0, H0, W0, 1, NB * H0 * (W0 / 16));

    // 4. q/k/v projections
    qkv_kernel<<<128, 256, 0, stream>>>(multif, monof, qkvf, q_b, k_b, v_b,
                                        Qhb, Khb, Vtb);

    // 5. flash attention (the hotspot: 2 x 8192 x 8192 x 32, never materialized)
    attn_kernel<<<128, 256, 0, stream>>>(Qhb, Khb, Vtb, moutb);

    // 6. rg conv2 fused with res + gamma * upsample(mono_out) -> NCHW f32 out
    conv3x3_fused_out<<<2048, 256, 0, stream>>>(rg1b, outp, wf + 5 * 9216,
        rg_b2, rg_g2, rg_be2, moutb, gamma);
}
